// ONOBlock_66228395704393
// MI455X (gfx1250) — matrix operand order repaired and checked by reading, not compiled, not run on hardware
//
#include <hip/hip_runtime.h>
#include <hip/hip_bf16.h>
#include <math.h>

// ---------------- problem constants ----------------
#define LB   8LL                 // batch
#define LNN  7225LL              // sequence length N
#define LD   256LL               // model dim D
#define LH   8LL                 // heads
#define LDH  32LL                // head dim
#define LPSI 64LL
#define LDF  1024LL
#define LM   (LB*LNN)            // 57800 tokens
#define LMP  57824LL             // M padded to mult of 32
#define LNP  7232LL              // N padded to mult of 32
#define LMB  (LB*LNP)            // 57856 batch-padded rows
#define LBH  (LB*LH)             // 64
#define TDMCH 226LL              // ceil(N/32) row chunks for TDM copy

typedef __bf16 bf16;
typedef __attribute__((ext_vector_type(16))) __bf16 v16bf;
typedef __attribute__((ext_vector_type(8)))  float  v8f;
typedef __attribute__((ext_vector_type(4)))  unsigned int v4u;
typedef __attribute__((ext_vector_type(4)))  int v4i;
typedef __attribute__((ext_vector_type(8)))  int v8i;

#if defined(__has_builtin)
#  if __has_builtin(__builtin_amdgcn_tensor_load_to_lds) && \
      __has_builtin(__builtin_amdgcn_tensor_store_from_lds)
#    define HAVE_TDM 1
#  endif
#endif
#ifndef HAVE_TDM
#  define HAVE_TDM 0
#endif

union Frag { v4u u[2]; v16bf v; };

__device__ __forceinline__ float gelu_exact(float x) {
    return 0.5f * x * (1.0f + erff(x * 0.70710678118654752f));
}

__device__ __forceinline__ void store_tile(v8f acc, int m0, int n,
    const float* bias, const float* R, float* Cf, bf16* Cb,
    size_t cfOff, size_t cbOff, int ldc,
    int maskMod, int maskVal, int gelu, float scale, int hi)
{
    float bn = bias ? bias[n] : 0.0f;
    #pragma unroll
    for (int r = 0; r < 8; ++r) {
        int m = m0 + hi * 8 + r;
        if ((m % maskMod) < maskVal) {
            float v = acc[r] * scale + bn;
            if (gelu) v = gelu_exact(v);
            size_t idx = (size_t)m * (size_t)ldc + (size_t)n;
            if (R)  v += R[cfOff + idx];
            if (Cf) Cf[cfOff + idx] = v;
            if (Cb) Cb[cbOff + idx] = (bf16)v;
        }
    }
}

// ---------------- generic batched bf16 WMMA GEMM, 32x32 C-tile per wave ----------------
// C[m,n] = scale * sum_k A[m,k] * Bt[n,k]   (+bias[n]) (gelu?) (+R[m,n])
// A: [Mtiles*32, lda] bf16 row-major ; Bt: [Ntiles*32, ldb] bf16 row-major (B^T)
// lda/ldb mult of 8, K mult of 32. store mask: (m % maskMod) < maskVal
__global__ void wmma_gemm(const bf16* A, long long sA, int lda,
                          const bf16* Bt, long long sB, int ldb,
                          const float* bias, const float* R,
                          float* Cf, long long sCf,
                          bf16* Cb, long long sCb,
                          int ldc, int Mtiles, int Ntiles, int K,
                          int maskMod, int maskVal, int gelu, float scale)
{
    int wave = (int)blockIdx.x * 8 + (int)(threadIdx.x >> 5);
    int lane = (int)(threadIdx.x & 31u);
    if (wave >= Mtiles * Ntiles) return;              // whole-wave uniform exit
    int tm = wave / Ntiles, tn = wave % Ntiles;
    long long batch = blockIdx.z;

    int lh = lane & 15, hi = lane >> 4;
    const bf16* aRow0 = A + batch * sA + (size_t)(tm * 32 + lh) * (size_t)lda;
    const bf16* aRow1 = aRow0 + (size_t)16 * (size_t)lda;
    const bf16* bRow0 = Bt + batch * sB + (size_t)(tn * 32 + lh) * (size_t)ldb;
    const bf16* bRow1 = bRow0 + (size_t)16 * (size_t)ldb;

    v8f acc00 = {}, acc01 = {}, acc10 = {}, acc11 = {};
    for (int k0 = 0; k0 < K; k0 += 32) {
        if (k0 + 128 < K) {                           // CDNA5 global_prefetch_b8
            __builtin_prefetch(aRow0 + k0 + 128, 0, 1);
            __builtin_prefetch(aRow1 + k0 + 128, 0, 1);
            __builtin_prefetch(bRow0 + k0 + 128, 0, 1);
            __builtin_prefetch(bRow1 + k0 + 128, 0, 1);
        }
        Frag fa0, fa1, fb0, fb1;
        const v4u* pa0 = (const v4u*)(aRow0 + k0 + hi * 8);
        const v4u* pa1 = (const v4u*)(aRow1 + k0 + hi * 8);
        fa0.u[0] = pa0[0]; fa0.u[1] = pa0[2];         // A: K = k0+hi*8, k0+16+hi*8
        fa1.u[0] = pa1[0]; fa1.u[1] = pa1[2];
        const v4u* pb0 = (const v4u*)(bRow0 + k0 + hi * 8);
        const v4u* pb1 = (const v4u*)(bRow1 + k0 + hi * 8);
        fb0.u[0] = pb0[0]; fb0.u[1] = pb0[2];         // B: K = k0+hi*8, k0+16+hi*8
        fb1.u[0] = pb1[0]; fb1.u[1] = pb1[2];
        acc00 = __builtin_amdgcn_wmma_f32_16x16x32_bf16(false, fa0.v, false, fb0.v, (short)0, acc00, false, false);
        acc01 = __builtin_amdgcn_wmma_f32_16x16x32_bf16(false, fa0.v, false, fb1.v, (short)0, acc01, false, false);
        acc10 = __builtin_amdgcn_wmma_f32_16x16x32_bf16(false, fa1.v, false, fb0.v, (short)0, acc10, false, false);
        acc11 = __builtin_amdgcn_wmma_f32_16x16x32_bf16(false, fa1.v, false, fb1.v, (short)0, acc11, false, false);
    }

    size_t cfOff = (size_t)(batch * sCf);
    size_t cbOff = (size_t)(batch * sCb);
    int m0 = tm * 32, n0 = tn * 32;
    store_tile(acc00, m0,      n0 + lh,      bias, R, Cf, Cb, cfOff, cbOff, ldc, maskMod, maskVal, gelu, scale, hi);
    store_tile(acc01, m0,      n0 + 16 + lh, bias, R, Cf, Cb, cfOff, cbOff, ldc, maskMod, maskVal, gelu, scale, hi);
    store_tile(acc10, m0 + 16, n0 + lh,      bias, R, Cf, Cb, cfOff, cbOff, ldc, maskMod, maskVal, gelu, scale, hi);
    store_tile(acc11, m0 + 16, n0 + 16 + lh, bias, R, Cf, Cb, cfOff, cbOff, ldc, maskMod, maskVal, gelu, scale, hi);
}

// ---------------- zero fill (b128 stores) ----------------
__global__ void zero_fill(v4u* p, long long n16)
{
    long long i = (long long)blockIdx.x * 256 + threadIdx.x;
    long long stride = (long long)gridDim.x * 256;
    v4u z = 0;
    for (; i < n16; i += stride) p[i] = z;
}

// ---------------- LayerNorm(D=256) -> bf16, zero for invalid rows ----------------
__global__ void ln_kernel(const float* X, const float* g, const float* bv,
                          bf16* out, long long rows, long long rowMod, long long rowVal)
{
    long long row = (long long)blockIdx.x * 8 + (threadIdx.x >> 5);
    int lane = (int)(threadIdx.x & 31u);
    if (row >= rows) return;
    bf16* o = out + row * 256 + lane;
    if ((row % rowMod) >= rowVal) {
        #pragma unroll
        for (int i = 0; i < 8; ++i) o[i * 32] = (bf16)0.0f;
        return;
    }
    const float* x = X + row * 256;
    float vals[8]; float s = 0.0f;
    #pragma unroll
    for (int i = 0; i < 8; ++i) { vals[i] = x[lane + i * 32]; s += vals[i]; }
    for (int m = 16; m; m >>= 1) s += __shfl_xor(s, m, 32);
    float mean = s * (1.0f / 256.0f);
    float v2 = 0.0f;
    #pragma unroll
    for (int i = 0; i < 8; ++i) { float d = vals[i] - mean; v2 += d * d; }
    for (int m = 16; m; m >>= 1) v2 += __shfl_xor(v2, m, 32);
    float inv = rsqrtf(v2 * (1.0f / 256.0f) + 1e-5f);
    #pragma unroll
    for (int i = 0; i < 8; ++i) {
        int c = lane + i * 32;
        o[i * 32] = (bf16)((vals[i] - mean) * inv * g[c] + bv[c]);
    }
}

// ---------------- q feature softmax: one wave per (token, head) ----------------
__global__ void softmax_q(const float* qkv, bf16* qb)
{
    long long w = (long long)blockIdx.x * 8 + (threadIdx.x >> 5);
    int lane = (int)(threadIdx.x & 31u);
    if (w >= LM * LH) return;
    long long m = w >> 3; int h = (int)(w & 7);
    float v = qkv[m * 768 + h * 32 + lane];
    float mx = v;
    for (int s = 16; s; s >>= 1) mx = fmaxf(mx, __shfl_xor(mx, s, 32));
    float e = __expf(v - mx);
    float sum = e;
    for (int s = 16; s; s >>= 1) sum += __shfl_xor(sum, s, 32);
    float q = e / sum * 0.17677669529663689f;     // * DH^-0.5
    long long b = m / LNN, n = m % LNN;
    qb[(((b * LH + h) * LNP) + n) * 32 + lane] = (bf16)q;
}

// ---------------- k sequence softmax: one block per (b,h,d) ----------------
__global__ void softmax_k(const float* qkv, bf16* kt)
{
    int col = (int)blockIdx.x;             // (b*8+h)*32+d in 0..2047
    int b = col >> 8, hd = col & 255;
    int tid = (int)threadIdx.x;
    __shared__ float red[256];
    const float* base = qkv + (long long)b * LNN * 768 + 256 + hd;
    float mx = -1e30f;
    for (long long n = tid; n < LNN; n += 256) mx = fmaxf(mx, base[n * 768]);
    red[tid] = mx; __syncthreads();
    for (int s = 128; s; s >>= 1) { if (tid < s) red[tid] = fmaxf(red[tid], red[tid + s]); __syncthreads(); }
    mx = red[0]; __syncthreads();
    float sum = 0.0f;
    for (long long n = tid; n < LNN; n += 256) sum += __expf(base[n * 768] - mx);
    red[tid] = sum; __syncthreads();
    for (int s = 128; s; s >>= 1) { if (tid < s) red[tid] += red[tid + s]; __syncthreads(); }
    float inv = 1.0f / red[0];
    bf16* o = kt + (long long)col * LNP;
    for (long long n = tid; n < LNN; n += 256)
        o[n] = (bf16)(__expf(base[n * 768] - mx) * inv);
}

// ---------------- v: [M,768] cols 512.. -> vt[bh][e][n] bf16 ----------------
__global__ void transpose_v(const float* qkv, bf16* vt)
{
    long long i = (long long)blockIdx.x * 256 + threadIdx.x;
    if (i >= LBH * 32 * LNN) return;
    long long col = i / LNN, n = i % LNN;
    long long b = col >> 8; int he = (int)(col & 255);
    vt[col * LNP + n] = (bf16)qkv[(b * LNN + n) * 768 + 512 + he];
}

// ---------------- ctx f32 [bh,32,32] -> ctxT bf16 [bh,e,d] ----------------
__global__ void t32(const float* ctx, bf16* ctxT)
{
    long long i = (long long)blockIdx.x * 256 + threadIdx.x;
    if (i >= LBH * 1024) return;
    long long bh = i >> 10; int de = (int)(i & 1023); int d = de >> 5, e = de & 31;
    ctxT[bh * 1024 + e * 32 + d] = (bf16)ctx[bh * 1024 + d * 32 + e];
}

// ---------------- attn [bh,NP,32] -> token-major bf16 [MP,256] ----------------
__global__ void merge_attn(const float* attnf, bf16* attnb)
{
    long long i = (long long)blockIdx.x * 256 + threadIdx.x;
    if (i >= LMP * 256) return;
    long long m = i >> 8; int c = (int)(i & 255);
    if (m >= LM) { attnb[i] = (bf16)0.0f; return; }
    int h = c >> 5, e = c & 31;
    long long b = m / LNN, n = m % LNN;
    attnb[i] = (bf16)attnf[(((b * LH + h) * LNP) + n) * 32 + e];
}

// ---------------- f32 -> bf16 with zero row padding ----------------
__global__ void cast_pad(const float* in, bf16* out, int ld, long long Mvalid, long long Mtotal)
{
    long long i = (long long)blockIdx.x * 256 + threadIdx.x;
    if (i >= Mtotal * ld) return;
    long long m = i / ld;
    out[i] = (m < Mvalid) ? (bf16)in[i] : (bf16)0.0f;
}

// ---------------- xpsi f32 [M,64] -> xpsib [MP,64] + xpsiT [64,MP] ----------------
__global__ void psi_prep(const float* xpsi, bf16* xpsib, bf16* xpsiT)
{
    long long i = (long long)blockIdx.x * 256 + threadIdx.x;
    if (i >= LMP * 64) return;
    long long m = i >> 6; int c = (int)(i & 63);
    bf16 v = (m < LM) ? (bf16)xpsi[m * 64 + c] : (bf16)0.0f;
    xpsib[i] = v;
    xpsiT[(long long)c * LMP + m] = v;
}

// ---------------- weight transpose-cast: W[K,Ncol] f32 -> Out[Ncol,K] bf16 ----------------
__global__ void wcastT(const float* W, bf16* out, int K, int Ncol)
{
    long long i = (long long)blockIdx.x * 256 + threadIdx.x;
    if (i >= (long long)K * Ncol) return;
    long long j = i / K, c = i % K;
    out[i] = (bf16)W[c * Ncol + j];
}

// ---------------- Cholesky(64x64) + lower-triangular inverse -> bf16 rows ----------------
__global__ void chol_inv(const float* cov, bf16* LinvB)
{
    __shared__ float L[64][64];
    __shared__ float Inv[64][64];
    int t = (int)threadIdx.x;                  // 0..63
    for (int j = 0; j < 64; ++j) L[t][j] = cov[t * 64 + j];
    __syncthreads();
    for (int j = 0; j < 64; ++j) {
        if (t == j) {
            float s = L[j][j];
            for (int q = 0; q < j; ++q) s -= L[j][q] * L[j][q];
            L[j][j] = sqrtf(fmaxf(s, 1e-20f));
        }
        __syncthreads();
        if (t > j) {
            float s = L[t][j];
            for (int q = 0; q < j; ++q) s -= L[t][q] * L[j][q];
            L[t][j] = s / L[j][j];
        }
        __syncthreads();
    }
    // thread t owns column t of Inv(L)
    Inv[t][t] = 1.0f / L[t][t];
    for (int i = t + 1; i < 64; ++i) {
        float s = 0.0f;
        for (int q = t; q < i; ++q) s += L[i][q] * Inv[q][t];
        Inv[i][t] = -s / L[i][i];
    }
    __syncthreads();
    // Bt layout for whitening: row j holds Linv[j,:]
    for (int c = 0; c < 64; ++c)
        LinvB[t * 64 + c] = (bf16)((c <= t) ? Inv[t][c] : 0.0f);
}

// ---------------- xw f32 [M,64] -> xwT [B,64,NP] + scaled xwsb [B,NP,64] ----------------
__global__ void xw_post(const float* xw, const float* mu, bf16* xwT, bf16* xwsb)
{
    long long i = (long long)blockIdx.x * 256 + threadIdx.x;
    if (i >= LM * 64) return;
    long long m = i >> 6; int p = (int)(i & 63);
    long long b = m / LNN, n = m % LNN;
    float v = xw[i];
    xwT[((b * 64) + p) * LNP + n] = (bf16)v;
    float muv = mu[p];
    float sp = (muv > 20.0f) ? muv : log1pf(__expf(muv));  // softplus
    xwsb[((b * LNP) + n) * 64 + p] = (bf16)(v * sp);
}

// ---------------- fx f32 [B,N,256] -> fxT bf16 [B,256,NP] ----------------
__global__ void fxT_prep(const float* fx, bf16* fxT)
{
    long long i = (long long)blockIdx.x * 256 + threadIdx.x;
    if (i >= LB * 256 * LNN) return;
    long long bd = i / LNN, n = i % LNN;
    long long b = bd >> 8; int d = (int)(bd & 255);
    fxT[bd * LNP + n] = (bf16)fx[((b * LNN) + n) * 256 + d];
}

// ---------------- ctx2 f32 [B,64,256] -> ctx2T bf16 [B,256,64] ----------------
__global__ void ctx2T_k(const float* ctx2, bf16* ctx2T)
{
    long long i = (long long)blockIdx.x * 256 + threadIdx.x;
    if (i >= LB * 64 * 256) return;
    long long b = i / (64 * 256); int pd = (int)(i % (64 * 256));
    int p = pd >> 8, d = pd & 255;
    ctx2T[(b * 256 + d) * 64 + p] = (bf16)ctx2[i];
}

// ---------------- TDM row-compaction: fx2 [B,NP,256] -> out [B,N,256] ----------------
// One 32-row x 256-col f32 tile per block; the Tensor Data Mover pulls the tile
// Global->LDS (6-arg clang-23 builtin), the block fans it out to d_out; a
// tensor_store_from_lds mirrors the tile to scratch to exercise the store path.
__global__ void tdm_compact_fx(const float* fx2, float* out, float* mirror)
{
    __shared__ float buf[32 * 256];            // 32 KB tile
    int tid = (int)threadIdx.x;
    long long blk = blockIdx.x;                // b * TDMCH + chunk
    long long b = blk / TDMCH, c = blk % TDMCH;
    long long n0 = c * 32;
    int rows = (int)(((LNN - n0) < 32) ? (LNN - n0) : 32);
    const float* src = fx2 + ((b * LNP + n0) << 8);

    if (tid < 32) {                            // wave 0 drives the TDM
        unsigned lds = (unsigned)(size_t)&buf[0];
        unsigned long long gsrc = (unsigned long long)(const void*)src;
        unsigned long long gdst = (unsigned long long)(void*)(mirror + ((b * LNN + n0) << 8));

        v4u g0l, g0s; v8i g1;
        g0l[0] = 1u;                                          // count=1, gather off
        g0l[1] = lds;                                         // lds_addr
        g0l[2] = (unsigned)(gsrc & 0xffffffffull);            // global_addr[31:0]
        g0l[3] = (unsigned)((gsrc >> 32) & 0x1ffffffull) | 0x80000000u;  // addr[56:32] | type=2
        g0s = g0l;
        g0s[2] = (unsigned)(gdst & 0xffffffffull);
        g0s[3] = (unsigned)((gdst >> 32) & 0x1ffffffull) | 0x80000000u;

        unsigned td0 = 256, td1 = (unsigned)rows;             // tensor dims (4B units)
        g1[0] = (int)(2u << 16);                              // data_size=4B
        g1[1] = (int)((td0 & 0xffffu) << 16);                 // tensor_dim0[15:0]
        g1[2] = (int)((td0 >> 16) | ((td1 & 0xffffu) << 16)); // dim0 hi | dim1 lo
        g1[3] = (int)((td1 >> 16) | (256u << 16));            // dim1 hi | tile_dim0=256
        g1[4] = (int)((unsigned)rows);                        // tile_dim1=rows, tile_dim2=0
        g1[5] = 256;                                          // tensor_dim0_stride
        g1[6] = 0;                                            // stride hi | dim1_stride lo
        g1[7] = 0;

#if HAVE_TDM
        v4i z4 = 0; v8i z8 = 0;
        __builtin_amdgcn_tensor_load_to_lds(g0l, g1, z4, z4, z8, 0);
        __builtin_amdgcn_s_wait_tensorcnt(0);                 // tile resident in LDS
        __builtin_amdgcn_tensor_store_from_lds(g0s, g1, z4, z4, z8, 0);
        __builtin_amdgcn_s_wait_tensorcnt(0);
#else
        (void)g0s;
        for (int i = tid; i < rows * 256; i += 32) buf[i] = src[i];
#endif
    }
    __syncthreads();
    long long total = (long long)rows * 256;
    float* dst = out + (b * LNN + n0) * 256;
    for (long long i = tid; i < total; i += 256) dst[i] = buf[i];
}

// =====================================================================
extern "C" void kernel_launch(void* const* d_in, const int* in_sizes, int n_in,
                              void* d_out, int out_size, void* d_ws, size_t ws_size,
                              hipStream_t stream)
{
    (void)in_sizes; (void)n_in; (void)out_size; (void)ws_size;
    const float* X    = (const float*)d_in[0];
    const float* FX   = (const float*)d_in[1];
    const float* ln1g = (const float*)d_in[2];
    const float* ln1b = (const float*)d_in[3];
    const float* Wq   = (const float*)d_in[4];
    const float* Wk   = (const float*)d_in[5];
    const float* Wv   = (const float*)d_in[6];
    const float* Wo   = (const float*)d_in[7];
    const float* bo   = (const float*)d_in[8];
    const float* ln2g = (const float*)d_in[9];
    const float* ln2b = (const float*)d_in[10];
    const float* mW1  = (const float*)d_in[11];
    const float* mb1  = (const float*)d_in[12];
    const float* mW2  = (const float*)d_in[13];
    const float* mb2  = (const float*)d_in[14];
    const float* pW1  = (const float*)d_in[15];
    const float* pb1  = (const float*)d_in[16];
    const float* pW2  = (const float*)d_in[17];
    const float* pb2  = (const float*)d_in[18];
    const float* mu   = (const float*)d_in[19];
    const float* ln3g = (const float*)d_in[20];
    const float* ln3b = (const float*)d_in[21];
    const float* m2W1 = (const float*)d_in[22];
    const float* m2b1 = (const float*)d_in[23];
    const float* m2W2 = (const float*)d_in[24];
    const float* m2b2 = (const float*)d_in[25];

    float* outX  = (float*)d_out;                    // x  [B,N,256]
    float* outFX = outX + LM * 256;                  // fx [B,N,256]

    // ----- workspace bump allocator (256B aligned) -----
    char* w = (char*)d_ws; size_t off = 0;
    auto alloc = [&](size_t bytes) -> char* {
        char* p = w + off;
        off = (off + bytes + 255) & ~(size_t)255;
        return p;
    };
    // zero-filled staging region (padded bf16 buffers)
    size_t zstart = off;
    bf16* qb    = (bf16*)alloc(LBH * LNP * 32 * 2);      // q' [bh,NP,32]
    bf16* kt    = (bf16*)alloc(LBH * 32 * LNP * 2);      // k'^T [bh,32,NP]
    bf16* vt    = (bf16*)alloc(LBH * 32 * LNP * 2);      // v^T  [bh,32,NP]
    bf16* h1g   = (bf16*)alloc(LMB * 1024 * 2);          // MLP hidden (shared MLP/MLP2)
    bf16* p1b   = (bf16*)alloc(LMP * 256 * 2);           // proj hidden
    bf16* xpsib = (bf16*)alloc(LMP * 64 * 2);            // x_  [MP,64]
    bf16* xpsiT = (bf16*)alloc(64 * LMP * 2);            // x_^T [64,MP]
    bf16* xwT   = (bf16*)alloc(LB * 64 * LNP * 2);       // x_w^T per batch
    bf16* xwsb  = (bf16*)alloc(LB * LNP * 64 * 2);       // x_w * softplus(mu)
    bf16* fxTb  = (bf16*)alloc(LB * 256 * LNP * 2);      // fx^T per batch
    size_t zend = off;
    // non-zeroed buffers
    bf16*  hb    = (bf16*)alloc(LMP * 256 * 2);          // LN outputs (reused)
    bf16*  hb3   = (bf16*)alloc(LMB * 256 * 2);          // LN3 output
    float* qkv   = (float*)alloc(LM * 768 * 4);
    float* ctxf  = (float*)alloc(LBH * 1024 * 4);
    bf16*  ctxT  = (bf16*)alloc(LBH * 1024 * 2);
    float* attnf = (float*)alloc(LBH * LNP * 32 * 4);
    bf16*  attnb = (bf16*)alloc(LMP * 256 * 2);
    bf16*  WqkvT = (bf16*)alloc(768 * 256 * 2);
    bf16*  WoT   = (bf16*)alloc(256 * 256 * 2);
    bf16*  W1T   = (bf16*)alloc(1024 * 256 * 2);
    bf16*  W2T   = (bf16*)alloc(256 * 1024 * 2);
    bf16*  pW1T  = (bf16*)alloc(256 * 256 * 2);
    bf16*  pW2T  = (bf16*)alloc(64 * 256 * 2);
    bf16*  m2W1T = (bf16*)alloc(1024 * 256 * 2);
    bf16*  m2W2T = (bf16*)alloc(256 * 1024 * 2);
    float* xpsi  = (float*)alloc(LM * 64 * 4);
    float* cov   = (float*)alloc(64 * 64 * 4);
    bf16*  LinvB = (bf16*)alloc(64 * 64 * 2);
    float* xw    = (float*)alloc(LM * 64 * 4);
    float* ctx2  = (float*)alloc(LB * 64 * 256 * 4);
    bf16*  ctx2T = (bf16*)alloc(LB * 256 * 64 * 2);
    float* fx1   = (float*)alloc(LMB * 256 * 4);
    float* fx2   = (float*)alloc(LMB * 256 * 4);
    float* mirr  = (float*)alloc(LMB * 256 * 4);         // TDM store mirror
    bf16*  xb    = (bf16*)alloc(LMP * 256 * 2);

    const int BIG = 1 << 30;
    auto gemm = [&](const bf16* A, long long sA, int lda,
                    const bf16* Bt, long long sB, int ldb,
                    const float* bias, const float* R,
                    float* Cf, long long sCf, bf16* Cb, long long sCb,
                    int ldc, int Mt, int Nt, int K,
                    int mMod, int mVal, int gl, float scale, int batches) {
        long long tiles = (long long)Mt * Nt;              // 32x32 tiles per wave
        dim3 grid((unsigned)((tiles + 7) / 8), 1u, (unsigned)batches);
        wmma_gemm<<<grid, 256, 0, stream>>>(A, sA, lda, Bt, sB, ldb, bias, R,
                                            Cf, sCf, Cb, sCb, ldc, Mt, Nt, K,
                                            mMod, mVal, gl, scale);
    };
    auto ew = [&](long long total) { return dim3((unsigned)((total + 255) / 256)); };

    // 0) zero padded staging region
    zero_fill<<<dim3(4096), 256, 0, stream>>>((v4u*)(w + zstart), (long long)(zend - zstart) / 16);

    // 1) weight transpose-casts
    wcastT<<<ew(256 * 256), 256, 0, stream>>>(Wq, WqkvT, 256, 256);
    wcastT<<<ew(256 * 256), 256, 0, stream>>>(Wk, WqkvT + 256 * 256, 256, 256);
    wcastT<<<ew(256 * 256), 256, 0, stream>>>(Wv, WqkvT + 512 * 256, 256, 256);
    wcastT<<<ew(256 * 256), 256, 0, stream>>>(Wo, WoT, 256, 256);
    wcastT<<<ew(256 * 1024), 256, 0, stream>>>(mW1, W1T, 256, 1024);
    wcastT<<<ew(1024 * 256), 256, 0, stream>>>(mW2, W2T, 1024, 256);
    wcastT<<<ew(256 * 256), 256, 0, stream>>>(pW1, pW1T, 256, 256);
    wcastT<<<ew(256 * 64), 256, 0, stream>>>(pW2, pW2T, 256, 64);
    wcastT<<<ew(256 * 1024), 256, 0, stream>>>(m2W1, m2W1T, 256, 1024);
    wcastT<<<ew(1024 * 256), 256, 0, stream>>>(m2W2, m2W2T, 1024, 256);

    // 2) LN1 -> hb ; qkv = hb @ [Wq|Wk|Wv]
    ln_kernel<<<dim3((unsigned)(LMP / 8)), 256, 0, stream>>>(X, ln1g, ln1b, hb, LMP, BIG, LM);
    gemm(hb, 0, 256, WqkvT, 0, 256, nullptr, nullptr, qkv, 0, nullptr, 0,
         768, (int)(LMP / 32), 24, 256, BIG, (int)LM, 0, 1.0f, 1);

    // 3) attention feature maps
    softmax_q<<<dim3((unsigned)LM), 256, 0, stream>>>(qkv, qb);
    softmax_k<<<dim3(2048), 256, 0, stream>>>(qkv, kt);
    transpose_v<<<ew(LBH * 32 * LNN), 256, 0, stream>>>(qkv, vt);

    // 4) ctx[bh] = k'^T @ v  (32x32, K=NP) ; attn[bh] = q' @ ctx
    gemm(kt, 32 * LNP, (int)LNP, vt, 32 * LNP, (int)LNP, nullptr, nullptr,
         ctxf, 1024, nullptr, 0, 32, 1, 1, (int)LNP, BIG, BIG, 0, 1.0f, (int)LBH);
    t32<<<ew(LBH * 1024), 256, 0, stream>>>(ctxf, ctxT);
    gemm(qb, LNP * 32, 32, ctxT, 1024, 32, nullptr, nullptr,
         attnf, LNP * 32, nullptr, 0, 32, (int)(LNP / 32), 1, 32, BIG, (int)LNN, 0, 1.0f, (int)LBH);
    merge_attn<<<ew(LMP * 256), 256, 0, stream>>>(attnf, attnb);

    // 5) x = attn @ Wo + bo + x
    gemm(attnb, 0, 256, WoT, 0, 256, bo, X, outX, 0, nullptr, 0,
         256, (int)(LMP / 32), 8, 256, BIG, (int)LM, 0, 1.0f, 1);

    // 6) MLP block: x += GELU(LN2(x) @ W1 + b1) @ W2 + b2
    ln_kernel<<<dim3((unsigned)(LMP / 8)), 256, 0, stream>>>(outX, ln2g, ln2b, hb, LMP, BIG, LM);
    gemm(hb, 0, 256, W1T, 0, 256, mb1, nullptr, nullptr, 0, h1g, 0,
         1024, (int)(LMP / 32), 32, 256, BIG, (int)LM, 1, 1.0f, 1);
    gemm(h1g, 0, 1024, W2T, 0, 1024, mb2, outX, outX, 0, nullptr, 0,
         256, (int)(LMP / 32), 8, 1024, BIG, (int)LM, 0, 1.0f, 1);

    // 7) projection: x_ = GELU(x @ pW1 + pb1) @ pW2 + pb2
    cast_pad<<<ew(LMP * 256), 256, 0, stream>>>(outX, xb, 256, LM, LMP);
    gemm(xb, 0, 256, pW1T, 0, 256, pb1, nullptr, nullptr, 0, p1b, 0,
         256, (int)(LMP / 32), 8, 256, BIG, (int)LM, 1, 1.0f, 1);
    gemm(p1b, 0, 256, pW2T, 0, 256, pb2, nullptr, xpsi, 0, nullptr, 0,
         64, (int)(LMP / 32), 2, 256, BIG, (int)LM, 0, 1.0f, 1);
    psi_prep<<<ew(LMP * 64), 256, 0, stream>>>(xpsi, xpsib, xpsiT);

    // 8) cov = x_^T x_ / (B*N) ; Cholesky + inverse ; x_w = x_ @ Linv^T
    gemm(xpsiT, 0, (int)LMP, xpsiT, 0, (int)LMP, nullptr, nullptr, cov, 0, nullptr, 0,
         64, 2, 2, (int)LMP, BIG, BIG, 0, 1.0f / (float)LM, 1);
    chol_inv<<<dim3(1), 64, 0, stream>>>(cov, LinvB);
    gemm(xpsib, 0, 64, LinvB, 0, 64, nullptr, nullptr, xw, 0, nullptr, 0,
         64, (int)(LMP / 32), 2, 64, BIG, (int)LM, 0, 1.0f, 1);
    xw_post<<<ew(LM * 64), 256, 0, stream>>>(xw, mu, xwT, xwsb);

    // 9) rank-64 operator on fx: ctx2[b] = x_w^T fx ; fx1[b] = (x_w*sp(mu)) @ ctx2
    fxT_prep<<<ew(LB * 256 * LNN), 256, 0, stream>>>(FX, fxTb);
    gemm(xwT, 64 * LNP, (int)LNP, fxTb, 256 * LNP, (int)LNP, nullptr, nullptr,
         ctx2, 64 * 256, nullptr, 0, 256, 2, 8, (int)LNP, BIG, BIG, 0, 1.0f, (int)LB);
    ctx2T_k<<<ew(LB * 64 * 256), 256, 0, stream>>>(ctx2, ctx2T);
    gemm(xwsb, LNP * 64, 64, ctx2T, 256 * 64, 64, nullptr, nullptr,
         fx1, LNP * 256, nullptr, 0, 256, (int)(LNP / 32), 8, 64,
         (int)LNP, (int)LNN, 0, 1.0f, (int)LB);

    // 10) fx = GELU(LN3(fx1) @ m2W1 + b1) @ m2W2 + b2
    ln_kernel<<<dim3((unsigned)(LMB / 8)), 256, 0, stream>>>(fx1, ln3g, ln3b, hb3, LMB, LNP, LNN);
    gemm(hb3, 0, 256, m2W1T, 0, 256, m2b1, nullptr, nullptr, 0, h1g, 0,
         1024, (int)(LMB / 32), 32, 256, (int)LNP, (int)LNN, 1, 1.0f, 1);
    gemm(h1g, 0, 1024, m2W2T, 0, 1024, m2b2, nullptr, fx2, 0, nullptr, 0,
         256, (int)(LMB / 32), 8, 1024, (int)LNP, (int)LNN, 0, 1.0f, 1);

    // 11) TDM-driven row compaction: fx2 -> d_out fx region
    tdm_compact_fx<<<dim3((unsigned)(LB * TDMCH)), 256, 0, stream>>>(fx2, outFX, mirr);
}
